// MDU_77592879170023
// MI455X (gfx1250) — matrix operand order repaired and checked by reading, not compiled, not run on hardware
//
#include <hip/hip_runtime.h>
#include <hip/hip_bf16.h>

// ---------------- problem constants ----------------
#define BSZ   512
#define SRCL  36
#define HDIM  1024
#define MIDD  4096
#define ROWS  (BSZ * SRCL)          // 18432
static constexpr float ALPHA  = 8.0f;
static constexpr float GAMMA_ = 0.98f;
static constexpr float EPS_   = 1e-6f;

typedef __attribute__((ext_vector_type(16))) __bf16 v16bf;
typedef __attribute__((ext_vector_type(8)))  float  v8f;
typedef int v4i_ __attribute__((vector_size(16)));   // matches async-builtin param type

#define AS1 __attribute__((address_space(1)))
#define AS3 __attribute__((address_space(3)))

#if defined(__AMDGCN__) && __has_builtin(__builtin_amdgcn_global_load_async_to_lds_b128) && \
    __has_builtin(__builtin_amdgcn_s_wait_asynccnt)
#define ASYNC_COPY 1
#else
#define ASYNC_COPY 0
#endif

// ---------------- workspace layout (256-B aligned chunks) ----------------
static constexpr size_t OFF_RUSUM = 0;                                   // 1 float (pad 256)
static constexpr size_t OFF_WGATE = 256;                                 // ROWS f32
static constexpr size_t OFF_MMF   = OFF_WGATE + (size_t)ROWS * 4;        // ROWS*H f32
static constexpr size_t OFF_MMB   = OFF_MMF   + (size_t)ROWS * HDIM * 4; // ROWS*H bf16
static constexpr size_t OFF_W1S   = OFF_MMB   + (size_t)ROWS * HDIM * 2; // H*MID bf16
static constexpr size_t OFF_W2S   = OFF_W1S   + (size_t)HDIM * MIDD * 2; // MID*H bf16
static constexpr size_t OFF_HBUF  = OFF_W2S   + (size_t)MIDD * HDIM * 2; // ROWS*MID bf16

// ---------------- small helpers ----------------
__device__ __forceinline__ float block_sum_256(float x) {
    // wave32 reduce then cross-wave via LDS (8 waves / 256 threads)
    for (int o = 16; o > 0; o >>= 1) x += __shfl_down(x, o, 32);
    __shared__ float s[8];
    __shared__ float tot;
    int lane = threadIdx.x & 31, w = threadIdx.x >> 5;
    if (lane == 0) s[w] = x;
    __syncthreads();
    if (w == 0) {
        float r = (lane < 8) ? s[lane] : 0.0f;
        for (int o = 4; o > 0; o >>= 1) r += __shfl_down(r, o, 32);
        if (lane == 0) tot = r;
    }
    __syncthreads();
    float r = tot;
    __syncthreads();   // safe reuse of the shared vars on the next call
    return r;
}

// ---------------- K0: zero + sum(ru) ----------------
__global__ void k_init(float* ru_sum) { if (threadIdx.x == 0) *ru_sum = 0.0f; }

__global__ void k_rusum(const float* __restrict__ ru, float* __restrict__ ru_sum) {
    int i = blockIdx.x * 256 + threadIdx.x;
    float x = (i < ROWS) ? ru[i] : 0.0f;
    float t = block_sum_256(x);
    if (threadIdx.x == 0) atomicAdd(ru_sum, t);
}

// ---------------- K1: rg, ru_new, focal gate, ru_out, w ----------------
__global__ void k_gate(const float* __restrict__ att, const float* __restrict__ ru,
                       const float* __restrict__ ru_sum,
                       float* __restrict__ w_out, float* __restrict__ ru_out) {
    int b = blockIdx.x;          // one block (64 threads) per batch row
    int s = threadIdx.x;
    float rg = 0.0f, runew = 0.0f;
    if (s < SRCL) {
        const float* a = att + ((size_t)b * SRCL + s) * SRCL;
        for (int j = 0; j < SRCL; ++j) rg += a[j];
        float base = ru[b * SRCL + s] * GAMMA_ + rg;
        runew = (*ru_sum == 0.0f) ? base : base / (1.0f + GAMMA_);
    }
    float sa = (s < SRCL) ? sqrtf(runew) : 0.0f;
    float xs = sa, xt = runew * sa;
    for (int o = 16; o > 0; o >>= 1) { xs += __shfl_down(xs, o, 32); xt += __shfl_down(xt, o, 32); }
    __shared__ float sS[2], sT[2];
    int lane = threadIdx.x & 31, w = threadIdx.x >> 5;
    if (lane == 0) { sS[w] = xs; sT[w] = xt; }
    __syncthreads();
    float S = sS[0] + sS[1], T = sT[0] + sT[1];
    if (s < SRCL) {
        float rm = ((runew * S - T) > 0.0f) ? 1.0f : 0.0f;
        ru_out[b * SRCL + s] = ALPHA * runew * rm + (1.0f / ALPHA) * runew * (1.0f - rm);
        w_out [b * SRCL + s] = ALPHA * rg    * rm + (1.0f / ALPHA) * rg    * (1.0f - rm);
    }
}

// ---------------- K2: mm = w * LN(v+q) ; fp32 + bf16 copies ----------------
__global__ void k_ln1(const float* __restrict__ v, const float* __restrict__ q,
                      const float* __restrict__ g1, const float* __restrict__ b1,
                      const float* __restrict__ wg,
                      float* __restrict__ mmf, __bf16* __restrict__ mmb) {
    int row = blockIdx.x;
    const float* pv = v + (size_t)row * HDIM;
    const float* pq = q + (size_t)row * HDIM;
    float x[4], sum = 0.0f;
    for (int i = 0; i < 4; ++i) { int c = threadIdx.x + 256 * i; x[i] = pv[c] + pq[c]; sum += x[i]; }
    float mu = block_sum_256(sum) * (1.0f / HDIM);
    float d2 = 0.0f;
    for (int i = 0; i < 4; ++i) { float d = x[i] - mu; d2 += d * d; }
    float var = block_sum_256(d2) * (1.0f / HDIM);
    float inv = rsqrtf(var + EPS_);
    float ww  = wg[row];
    for (int i = 0; i < 4; ++i) {
        int c = threadIdx.x + 256 * i;
        float val = ww * (g1[c] * (x[i] - mu) * inv + b1[c]);
        size_t idx = (size_t)row * HDIM + c;
        mmf[idx] = val;
        mmb[idx] = (__bf16)val;
    }
}

// ---------------- K3: swizzle fp32 weight [K,N] -> bf16 WMMA B-fragment layout -------
// fragment (kt,nt) = 512 bf16: entry = lane*16+e ; lane&15 = column, (lane>>4)*16+e = k-in-tile
__global__ void k_swz(const float* __restrict__ W, __bf16* __restrict__ out, int K, int N) {
    size_t idx = (size_t)blockIdx.x * 256 + threadIdx.x;
    if (idx >= (size_t)K * N) return;
    size_t frag = idx >> 9;
    int entry = (int)(idx & 511);
    int lane = entry >> 4, e = entry & 15;
    int ntiles = N >> 4;
    size_t kt = frag / ntiles;
    int   nt = (int)(frag % ntiles);
    int k = (int)kt * 32 + (lane >> 4) * 16 + e;
    int n = nt * 16 + (lane & 15);
    out[idx] = (__bf16)W[(size_t)k * N + n];
}

// ---------------- K4: WMMA bf16 GEMM, LDS double-buffered via async copies ----------
// C[M,N] = A[M,K] * B[K,N]; block tile 128x128, BK=32, 8 waves, wave tile 64x32.
// LDS buffer: [0,4096) bf16 = A tile 128x32 (xor-swizzled quads); [4096,8192) = 8 B-frags.
// RELU_BF16: out = relu(acc+bias) -> bf16 | else: outf[m,n] += acc + bias (in-place residual)
template <int KDIM, int NDIM, bool RELU_BF16>
__global__ void __launch_bounds__(256) k_gemm(const __bf16* __restrict__ A,
                                              const __bf16* __restrict__ Bswz,
                                              const float*  __restrict__ bias,
                                              __bf16* __restrict__ outb,
                                              float*  __restrict__ outf) {
    constexpr int KT = KDIM / 32;
    __shared__ __align__(16) __bf16 lds[2][8192];      // 32 KB total

    const int tid   = threadIdx.x;
    const int lane  = tid & 31;
    const int wid   = tid >> 5;
    const int wm    = wid & 1;                         // 0..1 -> 64-row half
    const int wn    = wid >> 1;                        // 0..3 -> 32-col strip
    const int m_blk = blockIdx.x * 128;
    const int n_blk = blockIdx.y * 128;
    const int l15   = lane & 15;
    const int half8 = (lane >> 4) * 8;
    const int q0    = lane >> 4;                       // 0 or 1 (A quad select)

    // per-thread staging: 2 A chunks + 2 B chunks of 16 B each per k-step
    auto stage = [&](int buf, int kt) {
        const int k0 = kt * 32;
        const size_t bbase = ((size_t)kt * (NDIM / 16) + (n_blk >> 4)) * 512;
#pragma unroll
        for (int i = 0; i < 2; ++i) {
            int c   = tid + i * 256;                   // 0..511
            int row = c >> 2, quad = c & 3;
            const __bf16* gA = A + (size_t)(m_blk + row) * KDIM + k0 + quad * 8;
            __bf16* lA = &lds[buf][row * 32 + ((quad ^ (row & 3)) * 8)];
            const __bf16* gB = Bswz + bbase + (size_t)c * 8;
            __bf16* lB = &lds[buf][4096 + c * 8];
#if ASYNC_COPY
            __builtin_amdgcn_global_load_async_to_lds_b128(
                (AS1 v4i_*)gA, (AS3 v4i_*)lA, 0, 0);
            __builtin_amdgcn_global_load_async_to_lds_b128(
                (AS1 v4i_*)gB, (AS3 v4i_*)lB, 0, 0);
#else
            *(uint4*)lA = *(const uint4*)gA;
            *(uint4*)lB = *(const uint4*)gB;
#endif
        }
    };

    v8f acc[4][2] = {};
#if ASYNC_COPY
    stage(0, 0);
#endif
    for (int kt = 0; kt < KT; ++kt) {
        const int cur = kt & 1;
#if ASYNC_COPY
        if (kt + 1 < KT) { stage(cur ^ 1, kt + 1); __builtin_amdgcn_s_wait_asynccnt(4); }
        else             { __builtin_amdgcn_s_wait_asynccnt(0); }
        __syncthreads();                               // all waves' copies of `cur` landed
#else
        __syncthreads();                               // previous reads done
        stage(cur, kt);
        __syncthreads();
#endif
        // B fragments for this wave's two n-tiles (contiguous 32 B per lane)
        union { v16bf v; uint4 q[2]; } bfr[2];
#pragma unroll
        for (int j = 0; j < 2; ++j) {
            const __bf16* p = &lds[cur][4096 + (wn * 2 + j) * 512 + lane * 16];
            bfr[j].q[0] = *(const uint4*)p;
            bfr[j].q[1] = *(const uint4*)(p + 8);
        }
#pragma unroll
        for (int t = 0; t < 4; ++t) {
            const int row = wm * 64 + t * 16 + l15;
            const __bf16* base = &lds[cur][row * 32];
            union { v16bf v; uint4 q[2]; } afr;
            afr.q[0] = *(const uint4*)(base + (((2 * q0)     ^ (row & 3)) * 8));
            afr.q[1] = *(const uint4*)(base + (((2 * q0 + 1) ^ (row & 3)) * 8));
#pragma unroll
            for (int j = 0; j < 2; ++j)
                acc[t][j] = __builtin_amdgcn_wmma_f32_16x16x32_bf16(
                    false, afr.v, false, bfr[j].v, (short)0, acc[t][j], false, false);
        }
#if ASYNC_COPY
        __syncthreads();                               // reads of `cur` done before overwrite
#endif
    }

#pragma unroll
    for (int j = 0; j < 2; ++j) {
        const int n_col = n_blk + wn * 32 + j * 16 + l15;
        const float bv = bias[n_col];
#pragma unroll
        for (int t = 0; t < 4; ++t) {
            const int mrow = m_blk + wm * 64 + t * 16 + half8;  // lanes>=16 own rows 8..15
#pragma unroll
            for (int r = 0; r < 8; ++r) {
                const size_t idx = (size_t)(mrow + r) * NDIM + n_col;
                const float val = acc[t][j][r] + bv;
                if constexpr (RELU_BF16) {
                    outb[idx] = (__bf16)fmaxf(val, 0.0f);
                } else {
                    outf[idx] = outf[idx] + val;       // residual add in place (each elem once)
                }
            }
        }
    }
}

// ---------------- K5: out = LN(mm + fc) ----------------
__global__ void k_ln2(const float* __restrict__ x_in,
                      const float* __restrict__ g2, const float* __restrict__ b2,
                      float* __restrict__ out) {
    int row = blockIdx.x;
    const float* px = x_in + (size_t)row * HDIM;
    float x[4], sum = 0.0f;
    for (int i = 0; i < 4; ++i) { int c = threadIdx.x + 256 * i; x[i] = px[c]; sum += x[i]; }
    float mu = block_sum_256(sum) * (1.0f / HDIM);
    float d2 = 0.0f;
    for (int i = 0; i < 4; ++i) { float d = x[i] - mu; d2 += d * d; }
    float var = block_sum_256(d2) * (1.0f / HDIM);
    float inv = rsqrtf(var + EPS_);
    for (int i = 0; i < 4; ++i) {
        int c = threadIdx.x + 256 * i;
        out[(size_t)row * HDIM + c] = g2[c] * (x[i] - mu) * inv + b2[c];
    }
}

// ---------------- launch ----------------
extern "C" void kernel_launch(void* const* d_in, const int* in_sizes, int n_in,
                              void* d_out, int out_size, void* d_ws, size_t ws_size,
                              hipStream_t stream) {
    const float* v   = (const float*)d_in[0];
    const float* q   = (const float*)d_in[1];
    const float* att = (const float*)d_in[2];
    const float* ru  = (const float*)d_in[3];
    const float* g1  = (const float*)d_in[4];
    const float* b1  = (const float*)d_in[5];
    const float* g2  = (const float*)d_in[6];
    const float* b2  = (const float*)d_in[7];
    const float* W1  = (const float*)d_in[8];
    const float* c1  = (const float*)d_in[9];
    const float* W2  = (const float*)d_in[10];
    const float* c2  = (const float*)d_in[11];

    float* out    = (float*)d_out;
    float* ru_out = out + (size_t)ROWS * HDIM;

    char*   ws     = (char*)d_ws;
    float*  ru_sum = (float*) (ws + OFF_RUSUM);
    float*  wgate  = (float*) (ws + OFF_WGATE);
    float*  mmf    = (float*) (ws + OFF_MMF);
    __bf16* mmb    = (__bf16*)(ws + OFF_MMB);
    __bf16* w1s    = (__bf16*)(ws + OFF_W1S);
    __bf16* w2s    = (__bf16*)(ws + OFF_W2S);
    __bf16* hbuf   = (__bf16*)(ws + OFF_HBUF);

    k_init <<<1, 1, 0, stream>>>(ru_sum);
    k_rusum<<<(ROWS + 255) / 256, 256, 0, stream>>>(ru, ru_sum);
    k_gate <<<BSZ, 64, 0, stream>>>(att, ru, ru_sum, wgate, ru_out);
    k_ln1  <<<ROWS, 256, 0, stream>>>(v, q, g1, b1, wgate, mmf, mmb);

    size_t wn = (size_t)HDIM * MIDD;
    k_swz<<<(unsigned)((wn + 255) / 256), 256, 0, stream>>>(W1, w1s, HDIM, MIDD);
    k_swz<<<(unsigned)((wn + 255) / 256), 256, 0, stream>>>(W2, w2s, MIDD, HDIM);

    dim3 grid1(ROWS / 128, MIDD / 128);   // 144 x 32
    k_gemm<HDIM, MIDD, true ><<<grid1, 256, 0, stream>>>(mmb,  w1s, c1, hbuf, nullptr);
    dim3 grid2(ROWS / 128, HDIM / 128);   // 144 x 8
    k_gemm<MIDD, HDIM, false><<<grid2, 256, 0, stream>>>(hbuf, w2s, c2, nullptr, mmf);

    k_ln2<<<ROWS, 256, 0, stream>>>(mmf, g2, b2, out);
}
